// Qwen3Attention_14491219657323
// MI455X (gfx1250) — compile-verified, matrix operand order checked by reading, not run on hardware
//
#include <hip/hip_runtime.h>

#define SEQ 4096
#define HID 2048
#define NH  16
#define NKV 8
#define HD  128

typedef __attribute__((ext_vector_type(16))) __bf16        v16bf;
typedef __attribute__((ext_vector_type(8)))  float         v8f;
typedef __attribute__((ext_vector_type(4)))  unsigned int  u32x4;
typedef __attribute__((ext_vector_type(2)))  unsigned int  u32x2;

union FragBF {
  u32x4 u[2];
  v16bf v;
};

static __device__ __forceinline__ unsigned short f2bf(float f) {
  unsigned int u = __float_as_uint(f);
  u += 0x7FFFu + ((u >> 16) & 1u);           // round-to-nearest-even
  return (unsigned short)(u >> 16);
}

static __device__ __forceinline__ v8f wmma_bf16(v16bf a, v16bf b, v8f c) {
  // (neg_a, A, neg_b, B, c_mod, C, reuse_a, reuse_b)
  return __builtin_amdgcn_wmma_f32_16x16x32_bf16(false, a, false, b, (short)0, c,
                                                 false, false);
}

// Async copy of 16 bytes global -> LDS (ASYNCcnt-tracked, no VGPR data path).
static __device__ __forceinline__ void async_g2l_b128(unsigned lds_off,
                                                      const void* gptr) {
  asm volatile("global_load_async_to_lds_b128 %0, %1, off"
               :: "v"(lds_off), "v"((unsigned long long)gptr)
               : "memory");
}

// ---------------------------------------------------------------------------
// Elementwise f32 -> bf16 (vectorized x4).  n4 = element_count / 4.
// ---------------------------------------------------------------------------
__global__ __launch_bounds__(256) void conv_bf16(
    const float* __restrict__ X, unsigned short* __restrict__ Y, int n4)
{
  const int i = blockIdx.x * 256 + threadIdx.x;
  if (i < n4) {
    const float4 v = ((const float4*)X)[i];
    u32x2 p;
    p.x = (unsigned)f2bf(v.x) | ((unsigned)f2bf(v.y) << 16);
    p.y = (unsigned)f2bf(v.z) | ((unsigned)f2bf(v.w) << 16);
    ((u32x2*)Y)[i] = p;
  }
}

// ---------------------------------------------------------------------------
// GEMM: C[M,N] = A[M,K] @ B[N,K]^T + bias[N]   (A,B bf16 row-major, K-contig)
// 128x128 block tile, 8 waves (2x4), 64x32 per wave, bf16 WMMA, f32 accum.
// Double-buffered LDS staging via GLOBAL_LOAD_ASYNC_TO_LDS_B128: tile t+1
// copies overlap WMMA on tile t; async loads complete in order, so
// s_wait_asynccnt 8 releases the current tile while 8 stay in flight.
// ---------------------------------------------------------------------------
__global__ __launch_bounds__(256) void gemm_bf(
    const unsigned short* __restrict__ A, const unsigned short* __restrict__ B,
    const float* __restrict__ bias, float* __restrict__ C,
    int M, int N, int K)
{
  __shared__ __align__(16) unsigned short As[2][128][72];
  __shared__ __align__(16) unsigned short Bs[2][128][72];

  const int tid   = threadIdx.x;
  const int lane  = tid & 31;
  const int wave  = tid >> 5;
  const int wm    = (wave >> 2) * 64;     // 0 / 64
  const int wn    = (wave & 3) * 32;      // 0 / 32 / 64 / 96
  const int mb    = blockIdx.y * 128;
  const int nb    = blockIdx.x * 128;
  const int lrow  = lane & 15;
  const int khalf = (lane >> 4) * 8;      // K sub-chunk select per ISA layout

  v8f acc[4][2];
#pragma unroll
  for (int i = 0; i < 4; ++i)
#pragma unroll
    for (int j = 0; j < 2; ++j) acc[i][j] = {};

  auto issue = [&](int buf, int k0) {
#pragma unroll
    for (int i = 0; i < 4; ++i) {
      const int ch = tid + 256 * i;
      const int r  = ch >> 3;             // 0..127
      const int c8 = (ch & 7) * 8;        // 0,8,...,56
      async_g2l_b128((unsigned)(size_t)&As[buf][r][c8],
                     A + (size_t)(mb + r) * K + (k0 + c8));
      async_g2l_b128((unsigned)(size_t)&Bs[buf][r][c8],
                     B + (size_t)(nb + r) * K + (k0 + c8));
    }
  };

  const int T = K / 64;
  issue(0, 0);
  for (int t = 0; t < T; ++t) {
    const int cur = t & 1;
    if (t + 1 < T) {
      issue(1 - cur, (t + 1) * 64);
      asm volatile("s_wait_asynccnt 0x8" ::: "memory");  // current tile landed
    } else {
      asm volatile("s_wait_asynccnt 0x0" ::: "memory");
    }
    __syncthreads();

#pragma unroll
    for (int kk = 0; kk < 64; kk += 32) {
      FragBF a[4], b[2];
#pragma unroll
      for (int i = 0; i < 4; ++i) {
        const unsigned short* p = &As[cur][wm + i * 16 + lrow][kk + khalf];
        a[i].u[0] = *(const u32x4*)p;
        a[i].u[1] = *(const u32x4*)(p + 16);
      }
#pragma unroll
      for (int j = 0; j < 2; ++j) {
        const unsigned short* p = &Bs[cur][wn + j * 16 + lrow][kk + khalf];
        b[j].u[0] = *(const u32x4*)p;
        b[j].u[1] = *(const u32x4*)(p + 16);
      }
#pragma unroll
      for (int i = 0; i < 4; ++i)
#pragma unroll
        for (int j = 0; j < 2; ++j)
          acc[i][j] = wmma_bf16(a[i].v, b[j].v, acc[i][j]);
    }
    __syncthreads();   // all waves done with buf[cur] before it is refilled
  }

#pragma unroll
  for (int j = 0; j < 2; ++j) {
    const int n   = nb + wn + j * 16 + lrow;
    const float bvv = bias[n];
#pragma unroll
    for (int i = 0; i < 4; ++i) {
      const int m0 = mb + wm + i * 16 + (lane >> 4) * 8;
#pragma unroll
      for (int r = 0; r < 8; ++r)
        C[(size_t)(m0 + r) * N + n] = acc[i][j][r] + bvv;
    }
  }
}

// ---------------------------------------------------------------------------
// Per-head RMSNorm + RoPE, f32 -> bf16.  One wave per (s,h) row of 128.
// Lane l owns d in {l, l+32, l+64, l+96}; RoPE pairs (d, d+64) stay in-lane.
// ---------------------------------------------------------------------------
__global__ __launch_bounds__(256) void rope_norm(
    const float* __restrict__ X, const int* __restrict__ positions,
    const float* __restrict__ nw, unsigned short* __restrict__ Y, int H)
{
  const int wave = threadIdx.x >> 5;
  const int lane = threadIdx.x & 31;
  const int row  = blockIdx.x * 8 + wave;         // row in [0, SEQ*H)
  const int s    = row / H;

  const float* x = X + (size_t)row * HD;
  float f0 = x[lane], f1 = x[lane + 32], f2 = x[lane + 64], f3 = x[lane + 96];

  float ss = f0 * f0 + f1 * f1 + f2 * f2 + f3 * f3;
#pragma unroll
  for (int off = 1; off < 32; off <<= 1) ss += __shfl_xor(ss, off, 32);
  const float inv = rsqrtf(ss * (1.0f / 128.0f) + 1e-6f);

  f0 *= inv * nw[lane];
  f1 *= inv * nw[lane + 32];
  f2 *= inv * nw[lane + 64];
  f3 *= inv * nw[lane + 96];

  const float pos = (float)positions[s];
  const float lf  = 13.815510557964274f * (1.0f / 64.0f);  // ln(1e6)/64
  const float a0  = pos * __expf(-lf * (float)lane);
  const float a1  = pos * __expf(-lf * (float)(lane + 32));
  const float c0 = __cosf(a0), s0 = __sinf(a0);
  const float c1 = __cosf(a1), s1 = __sinf(a1);

  unsigned short* y = Y + (size_t)row * HD;
  y[lane]      = f2bf(f0 * c0 - f2 * s0);
  y[lane + 32] = f2bf(f1 * c1 - f3 * s1);
  y[lane + 64] = f2bf(f2 * c0 + f0 * s0);
  y[lane + 96] = f2bf(f3 * c1 + f1 * s1);
}

// ---------------------------------------------------------------------------
// V: (S, NKV*HD) f32  ->  Vt: (NKV, HD, S) bf16  (so P@V B-frags are contiguous)
// ---------------------------------------------------------------------------
__global__ __launch_bounds__(256) void v_post(
    const float* __restrict__ V, unsigned short* __restrict__ Vt)
{
  const int idx = blockIdx.x * 256 + threadIdx.x;   // [0, SEQ*NKV*HD)
  const int s   = idx >> 10;
  const int hd  = idx & 1023;                        // hkv*128 + d
  Vt[(size_t)hd * SEQ + s] = f2bf(V[idx]);
}

// ---------------------------------------------------------------------------
// Causal flash attention, GQA.  1 wave per (head, 16-query tile).
// Scores: Q(16xD) @ K^T via 4x2 WMMA; online softmax in C-layout; P re-laid
// through LDS into A-format; P(16x32) @ V(32x128) via 8 WMMA.  Output bf16.
// ---------------------------------------------------------------------------
__global__ __launch_bounds__(32) void attn_fa(
    const unsigned short* __restrict__ Qb,   // (S, NH, HD) bf16
    const unsigned short* __restrict__ Kb,   // (S, NKV, HD) bf16
    const unsigned short* __restrict__ Vt,   // (NKV, HD, S) bf16
    unsigned short* __restrict__ O)          // (S, NH*HD) bf16
{
  __shared__ __align__(16) unsigned short P[16][40];

  const int h     = blockIdx.x;
  const int qb    = blockIdx.y * 16;
  const int hk    = h >> 1;                 // GQA: 2 q-heads per kv-head
  const int lane  = threadIdx.x;
  const int lrow  = lane & 15;
  const int half8 = (lane >> 4) * 8;
  const int khalf = half8;                  // same 0/8 pattern for K chunks

  // Q fragments for this 16-row tile (K = 128 split into 4 chunks of 32)
  FragBF q[4];
  {
    const unsigned short* qp = Qb + ((size_t)(qb + lrow) * NH + h) * HD;
#pragma unroll
    for (int i = 0; i < 4; ++i) {
      q[i].u[0] = *(const u32x4*)(qp + i * 32 + khalf);
      q[i].u[1] = *(const u32x4*)(qp + i * 32 + khalf + 16);
    }
  }

  float mrow[8], lsum[8];
  v8f o[8];
#pragma unroll
  for (int r = 0; r < 8; ++r) { mrow[r] = -1e30f; lsum[r] = 0.0f; }
#pragma unroll
  for (int nt = 0; nt < 8; ++nt) o[nt] = {};

  const float scale = 0.08838834764831845f;   // 1/sqrt(128)
  const int qlast = qb + 15;
  const unsigned short* vhead = Vt + (size_t)hk * HD * SEQ;

  for (int kb = 0; kb <= qlast; kb += 32) {
    // prefetch next K/V tile while this one computes
    if (kb + 32 <= qlast) {
      __builtin_prefetch(Kb + ((size_t)(kb + 32 + lrow) * NKV + hk) * HD, 0, 1);
      __builtin_prefetch(vhead + (size_t)(lrow * 8) * SEQ + (kb + 32), 0, 1);
    }

    // ---- scores: two 16-key N-tiles
    v8f sc0 = {}, sc1 = {};
    const unsigned short* kp0 = Kb + ((size_t)(kb + lrow) * NKV + hk) * HD;
    const unsigned short* kp1 = Kb + ((size_t)(kb + 16 + lrow) * NKV + hk) * HD;
#pragma unroll
    for (int i = 0; i < 4; ++i) {
      FragBF b0, b1;
      b0.u[0] = *(const u32x4*)(kp0 + i * 32 + khalf);
      b0.u[1] = *(const u32x4*)(kp0 + i * 32 + khalf + 16);
      b1.u[0] = *(const u32x4*)(kp1 + i * 32 + khalf);
      b1.u[1] = *(const u32x4*)(kp1 + i * 32 + khalf + 16);
      sc0 = wmma_bf16(q[i].v, b0.v, sc0);
      sc1 = wmma_bf16(q[i].v, b1.v, sc1);
    }

    // ---- mask + online softmax (row stats replicated across each 16-lane half)
    const int col0 = kb + lrow, col1 = col0 + 16;
    float fsc[8];
#pragma unroll
    for (int r = 0; r < 8; ++r) {
      const int rowg = qb + r + half8;
      float v0 = (col0 <= rowg) ? sc0[r] * scale : -1e30f;
      float v1 = (col1 <= rowg) ? sc1[r] * scale : -1e30f;
      float mx = fmaxf(v0, v1);
#pragma unroll
      for (int off = 1; off < 16; off <<= 1) mx = fmaxf(mx, __shfl_xor(mx, off, 32));
      mx = fmaxf(mx, mrow[r]);
      fsc[r]  = __expf(mrow[r] - mx);
      mrow[r] = mx;
      const float p0 = __expf(v0 - mx);
      const float p1 = __expf(v1 - mx);
      float rs = p0 + p1;
#pragma unroll
      for (int off = 1; off < 16; off <<= 1) rs += __shfl_xor(rs, off, 32);
      lsum[r] = lsum[r] * fsc[r] + rs;
      sc0[r] = p0;
      sc1[r] = p1;
    }
#pragma unroll
    for (int nt = 0; nt < 8; ++nt)
#pragma unroll
      for (int r = 0; r < 8; ++r) o[nt][r] *= fsc[r];

    // ---- P: C-layout -> LDS -> A-layout fragment (bf16)
#pragma unroll
    for (int r = 0; r < 8; ++r) {
      P[r + half8][lrow]      = f2bf(sc0[r]);
      P[r + half8][lrow + 16] = f2bf(sc1[r]);
    }
    __syncthreads();
    FragBF pa;
    {
      const unsigned short* pp = &P[lrow][khalf];
      pa.u[0] = *(const u32x4*)pp;
      pa.u[1] = *(const u32x4*)(pp + 16);
    }
    __syncthreads();

    // ---- O += P(16x32) @ V(32x128), 8 N-tiles over head dim
    const unsigned short* vbase = vhead + kb;
#pragma unroll
    for (int nt = 0; nt < 8; ++nt) {
      FragBF bv;
      const unsigned short* vp = vbase + (size_t)(nt * 16 + lrow) * SEQ + khalf;
      bv.u[0] = *(const u32x4*)vp;
      bv.u[1] = *(const u32x4*)(vp + 16);
      o[nt] = wmma_bf16(pa.v, bv.v, o[nt]);
    }
  }

  // ---- epilogue: normalize and store bf16 (consumed by output projection)
#pragma unroll
  for (int nt = 0; nt < 8; ++nt) {
    const int d = h * HD + nt * 16 + lrow;
#pragma unroll
    for (int r = 0; r < 8; ++r) {
      const int rowg = qb + r + half8;
      O[(size_t)rowg * (NH * HD) + d] = f2bf(o[nt][r] / lsum[r]);
    }
  }
}

// ---------------------------------------------------------------------------
extern "C" void kernel_launch(void* const* d_in, const int* in_sizes, int n_in,
                              void* d_out, int out_size, void* d_ws, size_t ws_size,
                              hipStream_t stream) {
  const int*   positions = (const int*)  d_in[0];
  const float* hidden    = (const float*)d_in[1];
  const float* wq  = (const float*)d_in[2];
  const float* bq  = (const float*)d_in[3];
  const float* wk  = (const float*)d_in[4];
  const float* bk  = (const float*)d_in[5];
  const float* wv  = (const float*)d_in[6];
  const float* bv  = (const float*)d_in[7];
  const float* wo  = (const float*)d_in[8];
  const float* bo  = (const float*)d_in[9];
  const float* qnw = (const float*)d_in[10];
  const float* knw = (const float*)d_in[11];
  float* out = (float*)d_out;

  char* ws = (char*)d_ws;
  float*          Qf   = (float*)(ws);                               // 32MB
  float*          Kf   = (float*)(ws + ((size_t)32 << 20));          // 16MB (reused: Vt)
  float*          Vf   = (float*)(ws + ((size_t)48 << 20));          // 16MB
  unsigned short* Qb   = (unsigned short*)(ws + ((size_t)64  << 20)); // 16MB
  unsigned short* Kb   = (unsigned short*)(ws + ((size_t)80  << 20)); //  8MB
  unsigned short* Hbf  = (unsigned short*)(ws + ((size_t)88  << 20)); // 16MB
  unsigned short* Wqb  = (unsigned short*)(ws + ((size_t)104 << 20)); //  8MB
  unsigned short* Wkb  = (unsigned short*)(ws + ((size_t)112 << 20)); //  4MB
  unsigned short* Wvb  = (unsigned short*)(ws + ((size_t)116 << 20)); //  4MB
  unsigned short* Wob  = (unsigned short*)(ws + ((size_t)120 << 20)); //  8MB
  unsigned short* Vt   = (unsigned short*)Kf;   // alias: Kf consumed before v_post
  unsigned short* attnB = (unsigned short*)Qf;  // alias: Qf consumed before attn

  const dim3 blk(256);
  // one-time bf16 conversions (activations + weights)
  conv_bf16<<<dim3(SEQ * HID / 4 / 256), blk, 0, stream>>>(hidden, Hbf, SEQ * HID / 4);
  conv_bf16<<<dim3(NH  * HD * HID / 4 / 256), blk, 0, stream>>>(wq, Wqb, NH  * HD * HID / 4);
  conv_bf16<<<dim3(NKV * HD * HID / 4 / 256), blk, 0, stream>>>(wk, Wkb, NKV * HD * HID / 4);
  conv_bf16<<<dim3(NKV * HD * HID / 4 / 256), blk, 0, stream>>>(wv, Wvb, NKV * HD * HID / 4);
  conv_bf16<<<dim3(HID * NH * HD / 4 / 256), blk, 0, stream>>>(wo, Wob, HID * NH * HD / 4);
  // QKV projections (bf16 WMMA, double-buffered async-LDS staging)
  gemm_bf<<<dim3(NH  * HD / 128, SEQ / 128), blk, 0, stream>>>(Hbf, Wqb, bq, Qf, SEQ, NH  * HD, HID);
  gemm_bf<<<dim3(NKV * HD / 128, SEQ / 128), blk, 0, stream>>>(Hbf, Wkb, bk, Kf, SEQ, NKV * HD, HID);
  gemm_bf<<<dim3(NKV * HD / 128, SEQ / 128), blk, 0, stream>>>(Hbf, Wvb, bv, Vf, SEQ, NKV * HD, HID);
  // per-head RMSNorm + RoPE -> bf16
  rope_norm<<<dim3(SEQ * NH  / 8), blk, 0, stream>>>(Qf, positions, qnw, Qb, NH);
  rope_norm<<<dim3(SEQ * NKV / 8), blk, 0, stream>>>(Kf, positions, knw, Kb, NKV);
  // V -> bf16, transposed (heads x dim x seq)
  v_post<<<dim3(SEQ * NKV * HD / 256), blk, 0, stream>>>(Vf, Vt);
  // causal GQA flash attention -> bf16
  attn_fa<<<dim3(NH, SEQ / 16), dim3(32), 0, stream>>>(Qb, Kb, Vt, attnB);
  // output projection (bf16 WMMA) -> f32 out
  gemm_bf<<<dim3(HID / 128, SEQ / 128), blk, 0, stream>>>(attnB, Wob, bo, out, SEQ, HID, NH * HD);
}
